// FwpLayer_18476949307445
// MI455X (gfx1250) — matrix operand order repaired and checked by reading, not compiled, not run on hardware
//
#include <hip/hip_runtime.h>
#include <hip/hip_bf16.h>
#include <math.h>

// ---------------------------------------------------------------------------
// Fast-weight RNN, restructured:
//   pre[t,f] = fast_in_t . W1[:,f] + b1[f] + t * fast_in_t . Bupd[:,f]
//            + sum_k fast_in[t,k] * ( Wupd[k*F+f, :] . cumH_ex[t] )
// Big term = per-k 16x16x192 bf16 WMMA GEMM, Wupd (194.5 MB) read exactly once.
// ---------------------------------------------------------------------------

typedef __bf16 bf16_t;
typedef bf16_t v16bf __attribute__((ext_vector_type(16)));
typedef float  v8f   __attribute__((ext_vector_type(8)));
typedef float  v2f   __attribute__((ext_vector_type(2)));

#define I_SZ 834
#define H_SZ 190
#define F_SZ 250
#define O_SZ 512
#define T_SZ 256
#define KIN  1024     // I + H
#define HP   192      // H padded to multiple of 32 (WMMA K)
#define FP   256      // F padded to multiple of 16
#define NKB  32       // k-blocks (k = 0..1023 split into 32 blocks of 32)

// ---------------- 1) sequential RNN scan (single workgroup) ----------------
__global__ __launch_bounds__(1024) void rnn_scan_kernel(
    const float* __restrict__ x, const float* __restrict__ h0,
    const float* __restrict__ Wrnn, const float* __restrict__ brnn,
    float* __restrict__ fast_in, bf16_t* __restrict__ cumH)
{
  __shared__ float hx[KIN];      // concat(h, x_t)
  __shared__ float h_sh[H_SZ];
  __shared__ float h_new[H_SZ];
  __shared__ float cum[HP];      // running sum of h_new (exclusive prefix)
  const int tid  = threadIdx.x;
  const int lane = tid & 31;
  const int wv   = tid >> 5;     // 32 waves

  if (tid < H_SZ) h_sh[tid] = h0[tid];
  if (tid < HP)   cum[tid]  = 0.f;
  __syncthreads();

  for (int t = 0; t < T_SZ; ++t) {
    // hx = concat(h, x_t)   (reference order for the RNN cell)
    hx[tid] = (tid < H_SZ) ? h_sh[tid] : x[t * I_SZ + (tid - H_SZ)];
    __syncthreads();

    // h_new = tanh(Wrnn @ hx + brnn): one wave per row, shuffle reduction
    for (int r = wv; r < H_SZ; r += 32) {
      const float* wr = Wrnn + r * KIN;
      float p = 0.f;
      for (int c = lane; c < KIN; c += 32) p += wr[c] * hx[c];
      for (int off = 16; off > 0; off >>= 1) p += __shfl_xor(p, off, 32);
      if (lane == 0) h_new[r] = tanhf(p + brnn[r]);
    }
    __syncthreads();

    // exclusive prefix (value BEFORE adding h_new_t), bf16, zero-padded to 192
    if (tid < HP) {
      cumH[t * HP + tid] = (bf16_t)cum[tid];
      if (tid < H_SZ) cum[tid] += h_new[tid];
    }
    // fast_in_t = concat(x_t, h_new_t)   (reference order for FastNn)
    fast_in[t * KIN + tid] = (tid < I_SZ) ? x[t * I_SZ + tid] : h_new[tid - I_SZ];
    if (tid < H_SZ) h_sh[tid] = h_new[tid];
    __syncthreads();
  }
}

// -------- 2) big contraction via bf16 WMMA, Wupd streamed once -------------
// grid: 512 WGs = 16 f-tiles x 32 k-blocks; 8 waves/WG, each wave owns 2 t-tiles.
// partial[kb][t][f] (no atomics -> deterministic); reduced in the epilogue.
__global__ __launch_bounds__(256) void term2_wmma_kernel(
    const float* __restrict__ Wupd, const float* __restrict__ fast_in,
    const bf16_t* __restrict__ cumH, float* __restrict__ partial)
{
  const int fb  = blockIdx.x & 15;
  const int kb  = blockIdx.x >> 4;
  const int f0  = fb * 16;
  const int lane = threadIdx.x & 31;
  const int wv   = threadIdx.x >> 5;
  const int n    = lane & 15;     // tile column (t) / A row offset
  const int hi   = lane >> 4;     // lane half
  const int t0a  = wv * 32;
  const int t0b  = t0a + 16;

  // Preload B (cumH) for both t-tiles: 6 K-chunks of 32, lives in VGPRs all loop.
  v16bf Ba[6], Bb[6];
#pragma unroll
  for (int c = 0; c < 6; ++c) {
    typedef bf16_t v8bf_ __attribute__((ext_vector_type(8)));
    const bf16_t* pa_ = cumH + (size_t)(t0a + n) * HP + c * 32 + hi * 8;
    const bf16_t* pb_ = cumH + (size_t)(t0b + n) * HP + c * 32 + hi * 8;
    Ba[c] = __builtin_shufflevector(*(const v8bf_*)pa_, *(const v8bf_*)(pa_ + 16), 0,1,2,3,4,5,6,7,8,9,10,11,12,13,14,15);
    Bb[c] = __builtin_shufflevector(*(const v8bf_*)pb_, *(const v8bf_*)(pb_ + 16), 0,1,2,3,4,5,6,7,8,9,10,11,12,13,14,15);
  }

  v8f acca = {0.f,0.f,0.f,0.f,0.f,0.f,0.f,0.f};
  v8f accb = {0.f,0.f,0.f,0.f,0.f,0.f,0.f,0.f};
  const float* fia = fast_in + (size_t)(t0a + n) * KIN;
  const float* fib = fast_in + (size_t)(t0b + n) * KIN;

  for (int kk = 0; kk < 32; ++kk) {
    const int k = kb * 32 + kk;
    long row = (long)k * F_SZ + f0 + n;          // f>=250 lanes clamp (columns
    if (row > 255999L) row = 255999L;            //  250..255 are discarded later)
    const float* rp = Wupd + row * 190L;

    // prefetch next k's row (gfx1250 global_prefetch path)
    long nrow = (long)(k + 1) * F_SZ + f0 + n;
    if (nrow > 255999L) nrow = 255999L;
    __builtin_prefetch(Wupd + nrow * 190L, 0, 1);

    // A tile: rows f0..f0+15, K=192 (fp32 -> bf16 on the fly, WMMA A layout:
    // lane half 'hi': elems 0..7 -> j0+8*hi+e ; elems 8..15 -> j0+16+8*hi+e)
    v16bf A[6];
#pragma unroll
    for (int c = 0; c < 5; ++c) {
      const float* p1 = rp + c * 32 + hi * 8;    // 8-byte aligned (190*4 % 8 == 0)
      const float* p2 = p1 + 16;
      v2f q0 = *(const v2f*)(p1),     q1 = *(const v2f*)(p1 + 2);
      v2f q2 = *(const v2f*)(p1 + 4), q3 = *(const v2f*)(p1 + 6);
      v2f r0 = *(const v2f*)(p2),     r1 = *(const v2f*)(p2 + 2);
      v2f r2 = *(const v2f*)(p2 + 4), r3 = *(const v2f*)(p2 + 6);
      v16bf a;
      a[0]=(bf16_t)q0[0]; a[1]=(bf16_t)q0[1]; a[2]=(bf16_t)q1[0]; a[3]=(bf16_t)q1[1];
      a[4]=(bf16_t)q2[0]; a[5]=(bf16_t)q2[1]; a[6]=(bf16_t)q3[0]; a[7]=(bf16_t)q3[1];
      a[8]=(bf16_t)r0[0]; a[9]=(bf16_t)r0[1]; a[10]=(bf16_t)r1[0]; a[11]=(bf16_t)r1[1];
      a[12]=(bf16_t)r2[0]; a[13]=(bf16_t)r2[1]; a[14]=(bf16_t)r3[0]; a[15]=(bf16_t)r3[1];
      A[c] = a;
    }
    { // chunk 5 (j0 = 160): second run crosses j=190 -> guarded scalar loads
      const float* p1 = rp + 160 + hi * 8;       // j 160..175, always valid
      v2f q0 = *(const v2f*)(p1),     q1 = *(const v2f*)(p1 + 2);
      v2f q2 = *(const v2f*)(p1 + 4), q3 = *(const v2f*)(p1 + 6);
      v16bf a;
      a[0]=(bf16_t)q0[0]; a[1]=(bf16_t)q0[1]; a[2]=(bf16_t)q1[0]; a[3]=(bf16_t)q1[1];
      a[4]=(bf16_t)q2[0]; a[5]=(bf16_t)q2[1]; a[6]=(bf16_t)q3[0]; a[7]=(bf16_t)q3[1];
      const int j2 = 176 + hi * 8;
#pragma unroll
      for (int e = 0; e < 8; ++e) {
        const int j = j2 + e;
        a[8 + e] = (bf16_t)((j < H_SZ) ? rp[j] : 0.f);   // B is 0 there anyway
      }
      A[5] = a;
    }

    // M_k tile = Wupd_k(16xH) @ cumH^T(Hx16), K=192 via 6 bf16 WMMAs per t-tile
    v8f Ca = {0.f,0.f,0.f,0.f,0.f,0.f,0.f,0.f};
    v8f Cb = {0.f,0.f,0.f,0.f,0.f,0.f,0.f,0.f};
#pragma unroll
    for (int c = 0; c < 6; ++c) {
      Ca = __builtin_amdgcn_wmma_f32_16x16x32_bf16(false, A[c], false, Ba[c],
                                                   (short)0, Ca, false, false);
      Cb = __builtin_amdgcn_wmma_f32_16x16x32_bf16(false, A[c], false, Bb[c],
                                                   (short)0, Cb, false, false);
    }
    // scale by fast_in[t, k] (per-lane scalar: column t = t0 + (lane&15))
    const float sa = fia[k];
    const float sb = fib[k];
#pragma unroll
    for (int i = 0; i < 8; ++i) { acca[i] += sa * Ca[i]; accb[i] += sb * Cb[i]; }
  }

  // C layout: VGPR i, lane -> row f0 + i + 8*hi, col t0 + (lane&15)
  float* pa = partial + (size_t)kb * (T_SZ * FP);
#pragma unroll
  for (int i = 0; i < 8; ++i) {
    const int fr = f0 + i + hi * 8;
    pa[(size_t)(t0a + n) * FP + fr] = acca[i];
    pa[(size_t)(t0b + n) * FP + fr] = accb[i];
  }
}

// ------------- 3) epilogue: fp32 small GEMMs + SELU + W2 -------------------
__global__ __launch_bounds__(256) void epilogue_kernel(
    const float* __restrict__ fast_in, const float* __restrict__ partial,
    const float* __restrict__ W1, const float* __restrict__ b1,
    const float* __restrict__ Bupd, const float* __restrict__ W2,
    const float* __restrict__ b2, float* __restrict__ out)
{
  const int t = blockIdx.x;
  __shared__ float fi[KIN];
  __shared__ float hact[FP];
  for (int i = threadIdx.x; i < KIN; i += 256) fi[i] = fast_in[(size_t)t * KIN + i];
  __syncthreads();

  const int f = threadIdx.x;
  float h = 0.f;
  if (f < F_SZ) {
    float a1 = 0.f, a3 = 0.f;
    for (int k = 0; k < KIN; ++k) {
      const float v = fi[k];
      a1 += v * W1[(size_t)k * F_SZ + f];
      a3 += v * Bupd[(size_t)k * F_SZ + f];
    }
    float t2 = 0.f;   // deterministic reduction of k-block partials
    for (int b = 0; b < NKB; ++b) t2 += partial[(size_t)b * (T_SZ * FP) + (size_t)t * FP + f];
    const float pre = a1 + b1[f] + t2 + (float)t * a3;
    const float kScale = 1.0507009873554805f, kAlpha = 1.6732632423543772f;
    h = kScale * (pre > 0.f ? pre : kAlpha * expm1f(pre));
  }
  if (f < FP) hact[f] = h;
  __syncthreads();

  for (int o = threadIdx.x; o < O_SZ; o += 256) {
    float s = b2[o];
    for (int ff = 0; ff < F_SZ; ++ff) s += hact[ff] * W2[(size_t)ff * O_SZ + o];
    out[(size_t)t * O_SZ + o] = s;
  }
}

extern "C" void kernel_launch(void* const* d_in, const int* in_sizes, int n_in,
                              void* d_out, int out_size, void* d_ws, size_t ws_size,
                              hipStream_t stream) {
  (void)in_sizes; (void)n_in; (void)out_size; (void)ws_size;
  const float* x    = (const float*)d_in[0];
  const float* h0   = (const float*)d_in[1];
  const float* Wrnn = (const float*)d_in[2];
  const float* brnn = (const float*)d_in[3];
  const float* Wupd = (const float*)d_in[4];
  const float* bupd = (const float*)d_in[5];
  const float* W1   = (const float*)d_in[6];
  const float* b1   = (const float*)d_in[7];
  const float* W2   = (const float*)d_in[8];
  const float* b2   = (const float*)d_in[9];
  float* out = (float*)d_out;

  // workspace: fast_in(1 MB) | cumH bf16(96 KB) | partial(8 MB)  ~ 9.4 MB
  char* ws = (char*)d_ws;
  float*  fast_in = (float*)ws;
  bf16_t* cumH    = (bf16_t*)(ws + (size_t)T_SZ * KIN * 4);
  float*  partial = (float*)(ws + (size_t)T_SZ * KIN * 4 + (size_t)T_SZ * HP * 2);

  rnn_scan_kernel<<<1, 1024, 0, stream>>>(x, h0, Wrnn, brnn, fast_in, cumH);
  term2_wmma_kernel<<<16 * NKB, 256, 0, stream>>>(Wupd, fast_in, cumH, partial);
  epilogue_kernel<<<T_SZ, 256, 0, stream>>>(fast_in, partial, W1, b1, bupd, W2, b2, out);
}
